// MyGRU_61375082660542
// MI455X (gfx1250) — matrix operand order repaired and checked by reading, not compiled, not run on hardware
//
#include <hip/hip_runtime.h>
#include <math.h>

// ---------------------------------------------------------------------------
// GRU cell, B=32768, I=H=512, f32 in/out, bf16 WMMA GEMM core (gfx1250).
// A-panels staged via GLOBAL_LOAD_ASYNC_TO_LDS_B128 (ASYNCcnt) into
// double-buffered LDS; fragments pulled with ds_load_b128.
// ---------------------------------------------------------------------------

#define BQ 32768
#define IQ 512
#define HQ 512

#define LDS_ROW    72                 // 64 k-cols + 8 bf16 pad (bank spread)
#define LDS_PANEL  (128 * LDS_ROW)    // one 128x64 bf16 chunk

typedef __attribute__((ext_vector_type(16))) __bf16 v16bf;
typedef __attribute__((ext_vector_type(8)))  __bf16 v8bf;
typedef __attribute__((ext_vector_type(8)))  float  v8f;
typedef __attribute__((ext_vector_type(4)))  float  v4f;

static __device__ __forceinline__ v16bf cat8(v8bf lo, v8bf hi) {
    return __builtin_shufflevector(lo, hi, 0,1,2,3,4,5,6,7,8,9,10,11,12,13,14,15);
}

// A fragment (16x32 bf16) from an LDS chunk (row stride LDS_ROW bf16).
// lane<16: row+l16, K = k32..k32+15 ; lane>=16: K = k32+16..k32+31.
static __device__ __forceinline__ v16bf lds_afrag(const __bf16* sp, int row,
                                                  int k32, int l16, int half) {
    const __bf16* p = sp + (row + l16) * LDS_ROW + (k32 + half * 16);
    v8bf lo = *(const v8bf*)p;
    v8bf hi = *(const v8bf*)(p + 8);
    return cat8(lo, hi);
}

// B fragment (32x16 bf16, K x N), B = W^T, W row-major [N x K]:
// lane<16: n0+l16, K=k0..k0+15 ; lane>=16: n0+l16, K=k0+16..k0+31.
static __device__ __forceinline__ v16bf load_bfrag(const __bf16* __restrict__ base,
                                                   int n0, int k0, int K,
                                                   int l16, int half) {
    const __bf16* p = base + (size_t)(n0 + l16) * K + (k0 + half * 16);
    return *(const v16bf*)p;            // 32B aligned -> 2x global_load_b128
}

static __device__ __forceinline__ v8f wmma_bf16(v16bf a, v16bf b, v8f c) {
    return __builtin_amdgcn_wmma_f32_16x16x32_bf16(
        false, a, false, b, (short)0, c, false, false);
}

static __device__ __forceinline__ float sigmoidf_(float t) {
    return 1.0f / (1.0f + __expf(-t));
}

// Issue one 128x64 bf16 chunk copy: 256 threads x 4 async b128 each.
// src row stride = 512 bf16; dest rows padded to LDS_ROW.
static __device__ __forceinline__ void stage_chunk(const __bf16* __restrict__ src,
                                                   int tile_m, int kc,
                                                   unsigned sbase) {
    #pragma unroll
    for (int i = 0; i < 4; ++i) {
        const int id = threadIdx.x + i * 256;        // 0..1023
        const int r  = id >> 3;                      // 0..127
        const int cc = id & 7;                       // 0..7 (8 bf16 = 16B each)
        const unsigned lo = sbase + (unsigned)(r * (LDS_ROW * 2) + cc * 16);
        const unsigned long long ga =
            (unsigned long long)(src + (size_t)(tile_m + r) * 512 + kc + cc * 8);
        asm volatile("global_load_async_to_lds_b128 %0, %1, off"
                     :: "v"(lo), "v"(ga) : "memory");
    }
}

// ---------------------------------------------------------------------------
// f32 -> bf16 conversion (8 elements / thread, vectorized)
// ---------------------------------------------------------------------------
__global__ void cvt_f32_bf16(const float* __restrict__ s, __bf16* __restrict__ d, int n) {
    int i = (blockIdx.x * blockDim.x + threadIdx.x) * 8;
    if (i < n) {   // all sizes are multiples of 8
        v4f f0 = *(const v4f*)(s + i);
        v4f f1 = *(const v4f*)(s + i + 4);
        v8bf o;
        o[0] = (__bf16)f0[0]; o[1] = (__bf16)f0[1];
        o[2] = (__bf16)f0[2]; o[3] = (__bf16)f0[3];
        o[4] = (__bf16)f1[0]; o[5] = (__bf16)f1[1];
        o[6] = (__bf16)f1[2]; o[7] = (__bf16)f1[3];
        *(v8bf*)(d + i) = o;
    }
}

// ---------------------------------------------------------------------------
// Stage 1: z = sigmoid(x WzT + h UzT + bz), r = sigmoid(x WrT + h UrT + br),
//          a = x WhT + bh,  hr = h * r (bf16).
// Tile: 128(M) x 64(N) per block, 8 waves as 4x2, each wave 2x2 16x16 frags.
// Chunks 0..7 stream x (12 WMMA / 32-k), chunks 8..15 stream h (8 WMMA).
// ---------------------------------------------------------------------------
__global__ __launch_bounds__(256) void gru_gates(
    const __bf16* __restrict__ xb, const __bf16* __restrict__ hb,
    const __bf16* __restrict__ Wz, const __bf16* __restrict__ Wr,
    const __bf16* __restrict__ Wh,
    const __bf16* __restrict__ Uz, const __bf16* __restrict__ Ur,
    const float*  __restrict__ Wz_b, const float* __restrict__ Uz_b,
    const float*  __restrict__ Wr_b, const float* __restrict__ Ur_b,
    const float*  __restrict__ Wh_b,
    const float*  __restrict__ h_f32,
    float* __restrict__ z_buf, float* __restrict__ a_buf,
    __bf16* __restrict__ hr_buf)
{
    __shared__ __bf16 smem[2][LDS_PANEL];

    const int lane = threadIdx.x & 31;
    const int wave = threadIdx.x >> 5;
    const int l16  = lane & 15;
    const int half = lane >> 4;
    const int wm   = wave >> 1;              // 0..3
    const int wn   = wave & 1;               // 0..1
    const int tile_m = blockIdx.x * 128;
    const int m0   = tile_m + wm * 32;
    const int n0   = blockIdx.y * 64 + wn * 32;

    v8f accz[2][2], accr[2][2], acch[2][2];
    #pragma unroll
    for (int i = 0; i < 2; ++i)
        #pragma unroll
        for (int j = 0; j < 2; ++j) {
            accz[i][j] = (v8f)0.0f;
            accr[i][j] = (v8f)0.0f;
            acch[i][j] = (v8f)0.0f;
        }

    stage_chunk(xb, tile_m, 0, (unsigned)(unsigned long long)(&smem[0][0]));

    for (int c = 0; c < 16; ++c) {
        if (c + 1 < 16) {
            const __bf16* nsrc = (c + 1 < 8) ? xb : hb;
            stage_chunk(nsrc, tile_m, ((c + 1) & 7) * 64,
                        (unsigned)(unsigned long long)(&smem[(c + 1) & 1][0]));
            asm volatile("s_wait_asynccnt 4" ::: "memory");  // chunk c done
        } else {
            asm volatile("s_wait_asynccnt 0" ::: "memory");
        }
        __syncthreads();

        const __bf16* sp = &smem[c & 1][0];
        const int kc = (c & 7) * 64;

        if (c < 8) {   // x phase: Wz, Wr, Wh
            #pragma unroll
            for (int k32 = 0; k32 < 64; k32 += 32) {
                v16bf a[2], bz[2], br[2], bh[2];
                #pragma unroll
                for (int mf = 0; mf < 2; ++mf)
                    a[mf] = lds_afrag(sp, wm * 32 + mf * 16, k32, l16, half);
                #pragma unroll
                for (int nf = 0; nf < 2; ++nf) {
                    bz[nf] = load_bfrag(Wz, n0 + nf * 16, kc + k32, IQ, l16, half);
                    br[nf] = load_bfrag(Wr, n0 + nf * 16, kc + k32, IQ, l16, half);
                    bh[nf] = load_bfrag(Wh, n0 + nf * 16, kc + k32, IQ, l16, half);
                }
                #pragma unroll
                for (int mf = 0; mf < 2; ++mf)
                    #pragma unroll
                    for (int nf = 0; nf < 2; ++nf) {
                        accz[mf][nf] = wmma_bf16(a[mf], bz[nf], accz[mf][nf]);
                        accr[mf][nf] = wmma_bf16(a[mf], br[nf], accr[mf][nf]);
                        acch[mf][nf] = wmma_bf16(a[mf], bh[nf], acch[mf][nf]);
                    }
            }
        } else {       // h phase: Uz, Ur
            #pragma unroll
            for (int k32 = 0; k32 < 64; k32 += 32) {
                v16bf a[2], bz[2], br[2];
                #pragma unroll
                for (int mf = 0; mf < 2; ++mf)
                    a[mf] = lds_afrag(sp, wm * 32 + mf * 16, k32, l16, half);
                #pragma unroll
                for (int nf = 0; nf < 2; ++nf) {
                    bz[nf] = load_bfrag(Uz, n0 + nf * 16, kc + k32, HQ, l16, half);
                    br[nf] = load_bfrag(Ur, n0 + nf * 16, kc + k32, HQ, l16, half);
                }
                #pragma unroll
                for (int mf = 0; mf < 2; ++mf)
                    #pragma unroll
                    for (int nf = 0; nf < 2; ++nf) {
                        accz[mf][nf] = wmma_bf16(a[mf], bz[nf], accz[mf][nf]);
                        accr[mf][nf] = wmma_bf16(a[mf], br[nf], accr[mf][nf]);
                    }
            }
        }
        __syncthreads();   // buffer reuse fence
    }

    // epilogue: C layout -> (m = v + 8*half, n = l16)
    #pragma unroll
    for (int nf = 0; nf < 2; ++nf) {
        const int n  = n0 + nf * 16 + l16;
        const float bzv = Wz_b[n] + Uz_b[n];
        const float brv = Wr_b[n] + Ur_b[n];
        const float bhv = Wh_b[n];
        #pragma unroll
        for (int mf = 0; mf < 2; ++mf) {
            #pragma unroll
            for (int v = 0; v < 8; ++v) {
                const int m = m0 + mf * 16 + v + half * 8;
                const size_t idx = (size_t)m * HQ + n;
                const float z = sigmoidf_(accz[mf][nf][v] + bzv);
                const float r = sigmoidf_(accr[mf][nf][v] + brv);
                z_buf[idx]  = z;
                a_buf[idx]  = acch[mf][nf][v] + bhv;
                hr_buf[idx] = (__bf16)(h_f32[idx] * r);
            }
        }
    }
}

// ---------------------------------------------------------------------------
// Stage 2: ht = tanh(a + hr UhT + bu);  h' = h + z*(ht - h) -> both outputs.
// ---------------------------------------------------------------------------
__global__ __launch_bounds__(256) void gru_cand(
    const __bf16* __restrict__ hr, const __bf16* __restrict__ Uh,
    const float*  __restrict__ Uh_b,
    const float*  __restrict__ a_buf, const float* __restrict__ z_buf,
    const float*  __restrict__ h_f32,
    float* __restrict__ out0, float* __restrict__ out1)
{
    __shared__ __bf16 smem[2][LDS_PANEL];

    const int lane = threadIdx.x & 31;
    const int wave = threadIdx.x >> 5;
    const int l16  = lane & 15;
    const int half = lane >> 4;
    const int wm   = wave >> 1;
    const int wn   = wave & 1;
    const int tile_m = blockIdx.x * 128;
    const int m0   = tile_m + wm * 32;
    const int n0   = blockIdx.y * 64 + wn * 32;

    v8f acc[2][2];
    #pragma unroll
    for (int i = 0; i < 2; ++i)
        #pragma unroll
        for (int j = 0; j < 2; ++j)
            acc[i][j] = (v8f)0.0f;

    stage_chunk(hr, tile_m, 0, (unsigned)(unsigned long long)(&smem[0][0]));

    for (int c = 0; c < 8; ++c) {
        if (c + 1 < 8) {
            stage_chunk(hr, tile_m, (c + 1) * 64,
                        (unsigned)(unsigned long long)(&smem[(c + 1) & 1][0]));
            asm volatile("s_wait_asynccnt 4" ::: "memory");
        } else {
            asm volatile("s_wait_asynccnt 0" ::: "memory");
        }
        __syncthreads();

        const __bf16* sp = &smem[c & 1][0];
        const int kc = c * 64;
        #pragma unroll
        for (int k32 = 0; k32 < 64; k32 += 32) {
            v16bf a[2], b[2];
            #pragma unroll
            for (int mf = 0; mf < 2; ++mf)
                a[mf] = lds_afrag(sp, wm * 32 + mf * 16, k32, l16, half);
            #pragma unroll
            for (int nf = 0; nf < 2; ++nf)
                b[nf] = load_bfrag(Uh, n0 + nf * 16, kc + k32, HQ, l16, half);
            #pragma unroll
            for (int mf = 0; mf < 2; ++mf)
                #pragma unroll
                for (int nf = 0; nf < 2; ++nf)
                    acc[mf][nf] = wmma_bf16(a[mf], b[nf], acc[mf][nf]);
        }
        __syncthreads();
    }

    #pragma unroll
    for (int nf = 0; nf < 2; ++nf) {
        const int n = n0 + nf * 16 + l16;
        const float bu = Uh_b[n];
        #pragma unroll
        for (int mf = 0; mf < 2; ++mf) {
            #pragma unroll
            for (int v = 0; v < 8; ++v) {
                const int m = m0 + mf * 16 + v + half * 8;
                const size_t idx = (size_t)m * HQ + n;
                const float ht = tanhf(acc[mf][nf][v] + a_buf[idx] + bu);
                const float ho = h_f32[idx];
                const float hn = ho + z_buf[idx] * (ht - ho);
                out0[idx] = hn;
                out1[idx] = hn;
            }
        }
    }
}

// ---------------------------------------------------------------------------
extern "C" void kernel_launch(void* const* d_in, const int* in_sizes, int n_in,
                              void* d_out, int out_size, void* d_ws, size_t ws_size,
                              hipStream_t stream) {
    const float* x    = (const float*)d_in[0];
    const float* h    = (const float*)d_in[1];
    const float* Wz_w = (const float*)d_in[2];
    const float* Wz_b = (const float*)d_in[3];
    const float* Wr_w = (const float*)d_in[4];
    const float* Wr_b = (const float*)d_in[5];
    const float* Wh_w = (const float*)d_in[6];
    const float* Wh_b = (const float*)d_in[7];
    const float* Uz_w = (const float*)d_in[8];
    const float* Uz_b = (const float*)d_in[9];
    const float* Ur_w = (const float*)d_in[10];
    const float* Ur_b = (const float*)d_in[11];
    const float* Uh_w = (const float*)d_in[12];
    const float* Uh_b = (const float*)d_in[13];

    char* ws = (char*)d_ws;
    size_t off = 0;
    auto take = [&](size_t bytes) -> char* {
        char* p = ws + off;
        off += (bytes + 255) & ~(size_t)255;
        return p;
    };
    __bf16* xb   = (__bf16*)take((size_t)BQ * IQ * 2);
    __bf16* hb   = (__bf16*)take((size_t)BQ * HQ * 2);
    __bf16* Wzb  = (__bf16*)take((size_t)HQ * IQ * 2);
    __bf16* Wrb  = (__bf16*)take((size_t)HQ * IQ * 2);
    __bf16* Whb  = (__bf16*)take((size_t)HQ * IQ * 2);
    __bf16* Uzb  = (__bf16*)take((size_t)HQ * HQ * 2);
    __bf16* Urb  = (__bf16*)take((size_t)HQ * HQ * 2);
    __bf16* Uhb  = (__bf16*)take((size_t)HQ * HQ * 2);
    float*  zbuf = (float*)take((size_t)BQ * HQ * 4);
    float*  abuf = (float*)take((size_t)BQ * HQ * 4);
    __bf16* hrb  = (__bf16*)take((size_t)BQ * HQ * 2);

    auto cvt = [&](const float* s, __bf16* d, size_t n) {
        const int th = 256;
        const int per = th * 8;
        int blocks = (int)((n + per - 1) / per);
        cvt_f32_bf16<<<blocks, th, 0, stream>>>(s, d, (int)n);
    };
    cvt(x,    xb,  (size_t)BQ * IQ);
    cvt(h,    hb,  (size_t)BQ * HQ);
    cvt(Wz_w, Wzb, (size_t)HQ * IQ);
    cvt(Wr_w, Wrb, (size_t)HQ * IQ);
    cvt(Wh_w, Whb, (size_t)HQ * IQ);
    cvt(Uz_w, Uzb, (size_t)HQ * HQ);
    cvt(Ur_w, Urb, (size_t)HQ * HQ);
    cvt(Uh_w, Uhb, (size_t)HQ * HQ);

    dim3 grid(BQ / 128, HQ / 64);
    dim3 blk(256);
    gru_gates<<<grid, blk, 0, stream>>>(
        xb, hb, Wzb, Wrb, Whb, Uzb, Urb,
        Wz_b, Uz_b, Wr_b, Ur_b, Wh_b,
        h, zbuf, abuf, hrb);

    float* out0 = (float*)d_out;
    float* out1 = out0 + (size_t)BQ * HQ;
    gru_cand<<<grid, blk, 0, stream>>>(
        hrb, Uhb, Uh_b, abuf, zbuf, h, out0, out1);
}
